// HierarchicalDecoder_81930796138498
// MI455X (gfx1250) — compile-verified, matrix operand order checked by reading, not run on hardware
//
#include <hip/hip_runtime.h>
#include <stdint.h>
#include <stddef.h>

typedef __attribute__((ext_vector_type(16))) __bf16 v16bf;
typedef __attribute__((ext_vector_type(8)))  __bf16 v8bf;
typedef __attribute__((ext_vector_type(8)))  float  v8f;

#define B_   1024
#define H_   512
#define T_   256
#define LAT_ 256
#define BT_  16    // batch rows per workgroup (one WMMA M-tile)

// ---- native f32 -> bf16 (v_cvt_pk_bf16_f32 class, RNE) ----
static __device__ __forceinline__ __bf16 f32_to_bf16(float f) {
  return (__bf16)f;
}

// fast activations: v_exp_f32 + v_rcp_f32 (TRANS pipe, co-executes with WMMA)
static __device__ __forceinline__ float sigm(float x) {
  return __builtin_amdgcn_rcpf(1.0f + __expf(-x));
}
static __device__ __forceinline__ float tanh_f(float x) {
  return 2.0f * __builtin_amdgcn_rcpf(1.0f + __expf(-2.0f * x)) - 1.0f;
}

// Load a 16x32 bf16 WMMA fragment (A: rows = M, or B: rows = N of a row-major
// (N,K) weight). Per CDNA5 layout: lanes 0-15 hold K chunks [k0,k0+8) and
// [k0+16,k0+24); lanes 16-31 hold [k0+8,k0+16) and [k0+24,k0+32).
static __device__ __forceinline__ v16bf load_frag(const __bf16* __restrict__ p0,
                                                  int ldk, int lane) {
  const int r    = lane & 15;
  const int koff = (lane & 16) ? 8 : 0;
  const __bf16* p = p0 + (size_t)r * ldk + koff;
  v8bf lo = *(const v8bf*)(p);
  v8bf hi = *(const v8bf*)(p + 16);
  return __builtin_shufflevector(lo, hi, 0,1,2,3,4,5,6,7,8,9,10,11,12,13,14,15);
}

// ---- weight pre-conversion kernels (run once per launch; weights L2-resident) ----
__global__ void cvt_bf16_kernel(const float* __restrict__ s, __bf16* __restrict__ d, int n) {
  for (int i = blockIdx.x * 256 + threadIdx.x; i < n; i += gridDim.x * 256)
    d[i] = f32_to_bf16(s[i]);
}
__global__ void bias_sum_kernel(const float* __restrict__ a, const float* __restrict__ b,
                                float* __restrict__ d, int n) {
  int i = blockIdx.x * 256 + threadIdx.x;
  if (i < n) d[i] = a[i] + b[i];
}

// ==================== persistent 2-layer LSTM decoder ====================
__global__ __launch_bounds__(256)
void hd_lstm_kernel(const float* __restrict__ zp,   const float* __restrict__ zs,
                    const float* __restrict__ zst,  const float* __restrict__ bproj,
                    const float* __restrict__ Wih1, // (2048,2) f32, x-term done in VALU
                    const float* __restrict__ bias1,// bih1+bhh1 (2048)
                    const float* __restrict__ bias2,// bih2+bhh2 (2048)
                    const float* __restrict__ bo1,  // (256)
                    const float* __restrict__ Wo2,  // (2,256) f32
                    const float* __restrict__ bo2,  // (2)
                    const __bf16* __restrict__ Wproj, // (1024,256) bf16
                    const __bf16* __restrict__ Whh1,  // (2048,512) bf16
                    const __bf16* __restrict__ Wih2,  // (2048,512) bf16
                    const __bf16* __restrict__ Whh2,  // (2048,512) bf16
                    const __bf16* __restrict__ Wo1,   // (256,512)  bf16
                    float* __restrict__ out)          // (B,T,2)
{
  __shared__ __align__(16) __bf16 hA1[16 * 512];   // h1 tile, row-major bf16
  __shared__ __align__(16) __bf16 hA2[16 * 512];   // h2 tile
  __shared__ __align__(16) __bf16 zA [16 * 256];   // latent tile
  __shared__ __align__(16) float  rbuf[16 * 256];  // relu(h2@Wo1^T+bo1)
  __shared__ float ybuf[16][2];
  __shared__ float ysum[16][2];
  __shared__ float Wo2s[2 * 256];

  const int wg   = blockIdx.x;        // 64 workgroups, 16 batch rows each
  const int tid  = threadIdx.x;       // 256 threads = 8 waves
  const int lane = tid & 31;
  const int wave = tid >> 5;
  const int row0 = wg * BT_;

  const int nlo   = lane & 15;        // column-in-tile for C/D layout
  const int rbase = (lane >> 4) * 8;  // row base for this lane half

  // ---- stage z = [z_prim | z_skill | z_style] as bf16 (16 x 256) ----
  for (int i = tid; i < 16 * 256; i += 256) {
    int r = i >> 8, c = i & 255;
    int gr = row0 + r;
    float v;
    if (c < 64)       v = zp [gr * 64  + c];
    else if (c < 128) v = zs [gr * 64  + (c - 64)];
    else              v = zst[gr * 128 + (c - 128)];
    zA[r * 256 + c] = f32_to_bf16(v);
  }
  for (int i = tid; i < 2 * 256; i += 256) Wo2s[i] = Wo2[i];
  if (tid < 32) ybuf[tid >> 1][tid & 1] = 0.0f;   // x0 = 0
  __syncthreads();

  // ---- init = z @ Wproj^T + bproj : wave owns h cols [wave*64,+64) and
  //      matching c cols [512+wave*64,+64) so recurrent state stays wave-local
  v8f c1[4], c2[4];
  {
    #pragma unroll
    for (int t = 0; t < 4; ++t) {
      const int nh = wave * 64 + t * 16;
      const int nc = 512 + nh;
      v8f acch = {}; v8f accc = {};
      #pragma unroll 1
      for (int kt = 0; kt < 8; ++kt) {
        const int k0 = kt * 32;
        v16bf a  = load_frag(&zA[k0], 256, lane);
        v16bf bh = load_frag(Wproj + (size_t)nh * 256 + k0, 256, lane);
        v16bf bc = load_frag(Wproj + (size_t)nc * 256 + k0, 256, lane);
        acch = __builtin_amdgcn_wmma_f32_16x16x32_bf16(false, a, false, bh,
                                                       (short)0, acch, false, false);
        accc = __builtin_amdgcn_wmma_f32_16x16x32_bf16(false, a, false, bc,
                                                       (short)0, accc, false, false);
      }
      const float bbh = bproj[nh + nlo];
      const float bbc = bproj[nc + nlo];
      const int   col = nh + nlo;
      #pragma unroll
      for (int e = 0; e < 8; ++e) {
        float hv = acch[e] + bbh;
        __bf16 hb = f32_to_bf16(hv);
        hA1[(rbase + e) * 512 + col] = hb;
        hA2[(rbase + e) * 512 + col] = hb;
        accc[e] = accc[e] + bbc;
      }
      c1[t] = accc;
      c2[t] = accc;   // both layers start from (h0, c0)
    }
  }
  __syncthreads();

  const v8f vzero = {};

  // ======================= time loop =======================
  for (int tstep = 0; tstep < T_; ++tstep) {
    v8f acc[16];

    // ---------- layer 1: gates = h1 @ Whh1^T (+ x@Wih1^T + b later) ----------
    #pragma unroll
    for (int a0 = 0; a0 < 16; ++a0) acc[a0] = vzero;
    #pragma unroll 1
    for (int kt = 0; kt < 16; ++kt) {
      const int k0 = kt * 32;
      v16bf a = load_frag(&hA1[k0], 512, lane);
      __builtin_prefetch(Whh1 + (size_t)(wave * 64) * 512 + k0 + 512, 0, 0);
      #pragma unroll
      for (int g = 0; g < 4; ++g) {
        #pragma unroll
        for (int t = 0; t < 4; ++t) {
          const int n0 = g * 512 + wave * 64 + t * 16;
          v16bf b = load_frag(Whh1 + (size_t)n0 * 512 + k0, 512, lane);
          acc[g * 4 + t] = __builtin_amdgcn_wmma_f32_16x16x32_bf16(
              false, a, false, b, (short)0, acc[g * 4 + t], false, false);
        }
      }
    }

    float x0v[8], x1v[8];
    #pragma unroll
    for (int e = 0; e < 8; ++e) {
      x0v[e] = ybuf[rbase + e][0];
      x1v[e] = ybuf[rbase + e][1];
    }
    __syncthreads();   // all reads of old hA1 done before overwrite

    #pragma unroll
    for (int t = 0; t < 4; ++t) {
      const int hcol = wave * 64 + t * 16 + nlo;
      float w0[4], w1[4], bs[4];
      #pragma unroll
      for (int g = 0; g < 4; ++g) {
        const int n = g * 512 + hcol;
        w0[g] = Wih1[n * 2 + 0];
        w1[g] = Wih1[n * 2 + 1];
        bs[g] = bias1[n];
      }
      v8f ci = c1[t];
      #pragma unroll
      for (int e = 0; e < 8; ++e) {
        float iv = acc[0 * 4 + t][e] + bs[0] + x0v[e] * w0[0] + x1v[e] * w1[0];
        float fv = acc[1 * 4 + t][e] + bs[1] + x0v[e] * w0[1] + x1v[e] * w1[1];
        float gv = acc[2 * 4 + t][e] + bs[2] + x0v[e] * w0[2] + x1v[e] * w1[2];
        float ov = acc[3 * 4 + t][e] + bs[3] + x0v[e] * w0[3] + x1v[e] * w1[3];
        float cn = sigm(fv) * ci[e] + sigm(iv) * tanh_f(gv);
        float hn = sigm(ov) * tanh_f(cn);
        ci[e] = cn;
        hA1[(rbase + e) * 512 + hcol] = f32_to_bf16(hn);
      }
      c1[t] = ci;
    }
    __syncthreads();

    // ---------- layer 2: gates = [h1,h2] @ [Wih2;Whh2]^T  (K = 1024) ----------
    #pragma unroll
    for (int a0 = 0; a0 < 16; ++a0) acc[a0] = vzero;
    #pragma unroll 1
    for (int kt = 0; kt < 32; ++kt) {
      const int k0 = (kt & 15) * 32;
      const __bf16* Abase = (kt < 16) ? &hA1[k0] : &hA2[k0];
      const __bf16* Wbase = (kt < 16) ? Wih2 : Whh2;
      v16bf a = load_frag(Abase, 512, lane);
      __builtin_prefetch(Wbase + (size_t)(wave * 64) * 512 + k0 + 512, 0, 0);
      #pragma unroll
      for (int g = 0; g < 4; ++g) {
        #pragma unroll
        for (int t = 0; t < 4; ++t) {
          const int n0 = g * 512 + wave * 64 + t * 16;
          v16bf b = load_frag(Wbase + (size_t)n0 * 512 + k0, 512, lane);
          acc[g * 4 + t] = __builtin_amdgcn_wmma_f32_16x16x32_bf16(
              false, a, false, b, (short)0, acc[g * 4 + t], false, false);
        }
      }
    }
    __syncthreads();   // reads of old hA2 done

    #pragma unroll
    for (int t = 0; t < 4; ++t) {
      const int hcol = wave * 64 + t * 16 + nlo;
      float bs[4];
      #pragma unroll
      for (int g = 0; g < 4; ++g) bs[g] = bias2[g * 512 + hcol];
      v8f ci = c2[t];
      #pragma unroll
      for (int e = 0; e < 8; ++e) {
        float iv = acc[0 * 4 + t][e] + bs[0];
        float fv = acc[1 * 4 + t][e] + bs[1];
        float gv = acc[2 * 4 + t][e] + bs[2];
        float ov = acc[3 * 4 + t][e] + bs[3];
        float cn = sigm(fv) * ci[e] + sigm(iv) * tanh_f(gv);
        float hn = sigm(ov) * tanh_f(cn);
        ci[e] = cn;
        hA2[(rbase + e) * 512 + hcol] = f32_to_bf16(hn);
      }
      c2[t] = ci;
    }
    __syncthreads();

    // ---------- output head: r = relu(h2 @ Wo1^T + bo1) ----------
    v8f ro[2];
    ro[0] = vzero; ro[1] = vzero;
    #pragma unroll 1
    for (int kt = 0; kt < 16; ++kt) {
      const int k0 = kt * 32;
      v16bf a = load_frag(&hA2[k0], 512, lane);
      #pragma unroll
      for (int t = 0; t < 2; ++t) {
        const int n0 = wave * 32 + t * 16;
        v16bf b = load_frag(Wo1 + (size_t)n0 * 512 + k0, 512, lane);
        ro[t] = __builtin_amdgcn_wmma_f32_16x16x32_bf16(
            false, a, false, b, (short)0, ro[t], false, false);
      }
    }
    #pragma unroll
    for (int t = 0; t < 2; ++t) {
      const int n  = wave * 32 + t * 16 + nlo;
      const float bb = bo1[n];
      #pragma unroll
      for (int e = 0; e < 8; ++e) {
        float v = ro[t][e] + bb;
        rbuf[(rbase + e) * 256 + n] = v > 0.0f ? v : 0.0f;
      }
    }
    if (tid < 32) ysum[tid >> 1][tid & 1] = 0.0f;
    __syncthreads();

    // y = r @ Wo2^T + bo2 : 256-thread partial sums + LDS float atomics
    {
      const int b  = tid >> 4;
      const int j  = (tid >> 3) & 1;
      const int sg = tid & 7;
      float s = 0.0f;
      #pragma unroll 1
      for (int n = sg * 32; n < sg * 32 + 32; ++n)
        s += rbuf[b * 256 + n] * Wo2s[j * 256 + n];
      atomicAdd(&ysum[b][j], s);   // ds_add_f32
    }
    __syncthreads();
    if (tid < 32) {
      const int b = tid >> 1, j = tid & 1;
      float y = ysum[b][j] + bo2[j];
      ybuf[b][j] = y;                                   // feedback x_{t+1}
      out[((size_t)(row0 + b) * T_ + tstep) * 2 + j] = y;
    }
    __syncthreads();
  }
}

// ==================== host launcher ====================
extern "C" void kernel_launch(void* const* d_in, const int* in_sizes, int n_in,
                              void* d_out, int out_size, void* d_ws, size_t ws_size,
                              hipStream_t stream) {
  (void)in_sizes; (void)n_in; (void)out_size; (void)ws_size;
  const float* zp    = (const float*)d_in[0];
  const float* zs    = (const float*)d_in[1];
  const float* zst   = (const float*)d_in[2];
  const float* Wproj = (const float*)d_in[3];
  const float* bproj = (const float*)d_in[4];
  const float* Wih1  = (const float*)d_in[5];
  const float* Whh1  = (const float*)d_in[6];
  const float* bih1  = (const float*)d_in[7];
  const float* bhh1  = (const float*)d_in[8];
  const float* Wih2  = (const float*)d_in[9];
  const float* Whh2  = (const float*)d_in[10];
  const float* bih2  = (const float*)d_in[11];
  const float* bhh2  = (const float*)d_in[12];
  const float* Wo1   = (const float*)d_in[13];
  const float* bo1   = (const float*)d_in[14];
  const float* Wo2   = (const float*)d_in[15];
  const float* bo2   = (const float*)d_in[16];

  // d_ws layout: bf16 weight copies, then f32 fused biases
  __bf16* wsb     = (__bf16*)d_ws;
  __bf16* Wproj_b = wsb;                                  // 1024*256
  __bf16* Whh1_b  = wsb + 262144;                         // 2048*512
  __bf16* Wih2_b  = wsb + 262144 + 1048576;               // 2048*512
  __bf16* Whh2_b  = wsb + 262144 + 2 * 1048576;           // 2048*512
  __bf16* Wo1_b   = wsb + 262144 + 3 * 1048576;           // 256*512
  float*  freg    = (float*)((char*)d_ws + 2 * (262144 + 3 * 1048576 + 131072));
  float*  bias1   = freg;          // 2048
  float*  bias2   = freg + 2048;   // 2048

  cvt_bf16_kernel<<<512, 256, 0, stream>>>(Wproj, Wproj_b, 262144);
  cvt_bf16_kernel<<<1024, 256, 0, stream>>>(Whh1, Whh1_b, 1048576);
  cvt_bf16_kernel<<<1024, 256, 0, stream>>>(Wih2, Wih2_b, 1048576);
  cvt_bf16_kernel<<<1024, 256, 0, stream>>>(Whh2, Whh2_b, 1048576);
  cvt_bf16_kernel<<<256, 256, 0, stream>>>(Wo1,  Wo1_b,  131072);
  bias_sum_kernel<<<8, 256, 0, stream>>>(bih1, bhh1, bias1, 2048);
  bias_sum_kernel<<<8, 256, 0, stream>>>(bih2, bhh2, bias2, 2048);

  hd_lstm_kernel<<<B_ / BT_, 256, 0, stream>>>(
      zp, zs, zst, bproj, Wih1, bias1, bias2, bo1, Wo2, bo2,
      Wproj_b, Whh1_b, Wih2_b, Whh2_b, Wo1_b, (float*)d_out);
}
